// DynamicVisualTokenizer_25271587569758
// MI455X (gfx1250) — compile-verified, matrix operand order checked by reading, not run on hardware
//
#include <hip/hip_runtime.h>
#include <hip/hip_bf16.h>
#include <math.h>

// ---------------- problem constants ----------------
static constexpr int B_   = 32;
static constexpr int N_   = 256;
static constexpr int C_   = 1408;
static constexpr int H_   = 16;
static constexpr int D_   = 88;
static constexpr int HID_ = 6144;
static constexpr int T_   = B_ * N_;        // 8192 tokens
static constexpr int C3_  = 3 * C_;         // 4224
static constexpr float SCALE_ = 0.10660035817780522f; // 88^-0.5
static constexpr float EPS_   = 1e-6f;
static constexpr float EPSN_  = 1e-6f / 256.0f;
static constexpr float LN_EPS_ = 1e-5f;

// LDS row pitch for staged K/V (90 = conflict-free lane stride, gcd(26,64)=2)
static constexpr int KVP_ = 90;
static constexpr int ATTN_SMEM_FLOATS_ = 2 * N_ * KVP_ + 8 * 96 + 8 * N_ + N_;
static constexpr size_t ATTN_SMEM_BYTES_ = (size_t)ATTN_SMEM_FLOATS_ * 4; // 196608 B <= 320KB/WGP

// ---------------- WMMA vector types ----------------
typedef __attribute__((ext_vector_type(16))) __bf16 v16bf;
typedef __attribute__((ext_vector_type(8)))  __bf16 v8bf;
typedef __attribute__((ext_vector_type(8)))  float  v8f;

union FragAB { v16bf v; v8bf h[2]; };

__device__ __forceinline__ __bf16 f2bf(float f) {
  unsigned u = __builtin_bit_cast(unsigned, f);
  u += 0x7FFFu + ((u >> 16) & 1u);            // round-to-nearest-even
  unsigned short hs = (unsigned short)(u >> 16);
  return __builtin_bit_cast(__bf16, hs);
}

// ---------------- utility kernels ----------------
__global__ __launch_bounds__(256) void copy_f32(const float* __restrict__ s,
                                                float* __restrict__ d, size_t n) {
  size_t i = (size_t)blockIdx.x * blockDim.x + threadIdx.x;
  size_t st = (size_t)gridDim.x * blockDim.x;
  for (; i < n; i += st) d[i] = s[i];
}

__global__ __launch_bounds__(256) void cvt_f32_bf16(const float* __restrict__ s,
                                                    __bf16* __restrict__ d, size_t n) {
  size_t i = (size_t)blockIdx.x * blockDim.x + threadIdx.x;
  size_t st = (size_t)gridDim.x * blockDim.x;
  for (; i < n; i += st) d[i] = f2bf(s[i]);
}

// ---------------- LayerNorm: one 256-thread block per row of C=1408 ----------------
template<bool BF16OUT>
__global__ __launch_bounds__(256)
void ln_kernel(const float* __restrict__ x, const float* __restrict__ w,
               const float* __restrict__ bb, float* __restrict__ outF,
               __bf16* __restrict__ outB) {
  __shared__ float red[256];
  const int row = blockIdx.x;
  const int tid = threadIdx.x;
  const float* xr = x + (size_t)row * C_;

  float vals[6];
  int cnt = 0;
  float s = 0.f;
  for (int d = tid; d < C_; d += 256) { float v = xr[d]; vals[cnt++] = v; s += v; }
  red[tid] = s; __syncthreads();
  for (int st = 128; st > 0; st >>= 1) { if (tid < st) red[tid] += red[tid + st]; __syncthreads(); }
  const float mean = red[0] * (1.0f / C_);
  __syncthreads();

  float s2 = 0.f;
  for (int i = 0; i < cnt; i++) { float d0 = vals[i] - mean; s2 += d0 * d0; }
  red[tid] = s2; __syncthreads();
  for (int st = 128; st > 0; st >>= 1) { if (tid < st) red[tid] += red[tid + st]; __syncthreads(); }
  const float rstd = rsqrtf(red[0] * (1.0f / C_) + LN_EPS_);

  int i2 = 0;
  for (int d = tid; d < C_; d += 256) {
    float y = (vals[i2++] - mean) * rstd * w[d] + bb[d];
    if (BF16OUT) outB[(size_t)row * C_ + d] = f2bf(y);
    else         outF[(size_t)row * C_ + d] = y;
  }
}

// ---------------- bf16 WMMA GEMM:  out[M,Nout] = A[M,K] x W[Nout,K]^T + bias ----------------
// Block tile 128x128 (8 waves, each 32x64 -> 8 WMMAs per 12 fragment loads).
// EPI 0: outF = v      EPI 1: outF += v (residual)      EPI 2: outH = bf16(gelu(v))
template<int EPI>
__global__ __launch_bounds__(256)
void gemm_bf16_wmma(const __bf16* __restrict__ A, const __bf16* __restrict__ W,
                    const float* __restrict__ bias,
                    float* __restrict__ outF, __bf16* __restrict__ outH,
                    int M, int Nout, int K) {
  const int lane = threadIdx.x & 31;
  const int wave = threadIdx.x >> 5;
  const int wm = wave & 3;          // 4 waves along M (32 rows each) -> 128
  const int wn = wave >> 2;         // 2 waves along N (64 cols each) -> 128
  const int m0 = blockIdx.y * 128 + wm * 32;
  const int n0 = blockIdx.x * 128 + wn * 64;
  const int half = lane >> 4;       // K-half select per CDNA5 16-bit A/B layout
  const int lrow = lane & 15;

  v8f acc[2][4] = {};

  const __bf16* ap0 = A + (size_t)(m0 + lrow)      * K + half * 8;
  const __bf16* ap1 = A + (size_t)(m0 + 16 + lrow) * K + half * 8;
  const __bf16* bp0 = W + (size_t)(n0 + lrow)      * K + half * 8;
  const __bf16* bp1 = W + (size_t)(n0 + 16 + lrow) * K + half * 8;
  const __bf16* bp2 = W + (size_t)(n0 + 32 + lrow) * K + half * 8;
  const __bf16* bp3 = W + (size_t)(n0 + 48 + lrow) * K + half * 8;

  for (int k = 0; k < K; k += 32) {
    if (k + 128 < K) {               // gfx1250 global_prefetch_b8, 4 iterations ahead
      __builtin_prefetch((const void*)(ap0 + k + 128), 0, 1);
      __builtin_prefetch((const void*)(bp0 + k + 128), 0, 1);
    }
    FragAB a0, a1, b0, b1, b2, b3;
    a0.h[0] = *(const v8bf*)(ap0 + k);  a0.h[1] = *(const v8bf*)(ap0 + k + 16);
    a1.h[0] = *(const v8bf*)(ap1 + k);  a1.h[1] = *(const v8bf*)(ap1 + k + 16);
    b0.h[0] = *(const v8bf*)(bp0 + k);  b0.h[1] = *(const v8bf*)(bp0 + k + 16);
    b1.h[0] = *(const v8bf*)(bp1 + k);  b1.h[1] = *(const v8bf*)(bp1 + k + 16);
    b2.h[0] = *(const v8bf*)(bp2 + k);  b2.h[1] = *(const v8bf*)(bp2 + k + 16);
    b3.h[0] = *(const v8bf*)(bp3 + k);  b3.h[1] = *(const v8bf*)(bp3 + k + 16);

    acc[0][0] = __builtin_amdgcn_wmma_f32_16x16x32_bf16(false, a0.v, false, b0.v,
                                                        (short)0, acc[0][0], false, false);
    acc[0][1] = __builtin_amdgcn_wmma_f32_16x16x32_bf16(false, a0.v, false, b1.v,
                                                        (short)0, acc[0][1], false, false);
    acc[0][2] = __builtin_amdgcn_wmma_f32_16x16x32_bf16(false, a0.v, false, b2.v,
                                                        (short)0, acc[0][2], false, false);
    acc[0][3] = __builtin_amdgcn_wmma_f32_16x16x32_bf16(false, a0.v, false, b3.v,
                                                        (short)0, acc[0][3], false, false);
    acc[1][0] = __builtin_amdgcn_wmma_f32_16x16x32_bf16(false, a1.v, false, b0.v,
                                                        (short)0, acc[1][0], false, false);
    acc[1][1] = __builtin_amdgcn_wmma_f32_16x16x32_bf16(false, a1.v, false, b1.v,
                                                        (short)0, acc[1][1], false, false);
    acc[1][2] = __builtin_amdgcn_wmma_f32_16x16x32_bf16(false, a1.v, false, b2.v,
                                                        (short)0, acc[1][2], false, false);
    acc[1][3] = __builtin_amdgcn_wmma_f32_16x16x32_bf16(false, a1.v, false, b3.v,
                                                        (short)0, acc[1][3], false, false);
  }

  // C/D layout: lanes 0-15 -> M = r, N = lane; lanes 16-31 -> M = 8+r, N = lane-16
#pragma unroll
  for (int mi = 0; mi < 2; mi++) {
#pragma unroll
    for (int ni = 0; ni < 4; ni++) {
      const int col = n0 + ni * 16 + lrow;
      const float bv = bias[col];
#pragma unroll
      for (int r = 0; r < 8; r++) {
        const int row = m0 + mi * 16 + half * 8 + r;
        float v = acc[mi][ni][r] + bv;
        const size_t idx = (size_t)row * Nout + col;
        if (EPI == 0)      outF[idx] = v;
        else if (EPI == 1) outF[idx] += v;
        else               outH[idx] = f2bf(0.5f * v * (1.0f + erff(v * 0.70710678118f)));
      }
    }
  }
}

// ---------------- policy attention (causal or cross) ----------------
// One block per (b,h). Full K and V heads (256x88 f32) staged in LDS using the
// MI455X 320KB/WGP LDS (192KB used). 8 waves x 32 query iterations.
__global__ __launch_bounds__(256)
void attn_kernel(const float* __restrict__ Qb, const float* __restrict__ Kb,
                 const float* __restrict__ Vb, int ld,
                 const float* __restrict__ policy,
                 __bf16* __restrict__ outbf, int causal) {
  extern __shared__ float smem[];
  float* kt = smem;                    // [256][90]
  float* vt = kt + N_ * KVP_;          // [256][90]
  float* qw = vt + N_ * KVP_;          // [8][96]
  float* ww = qw + 8 * 96;             // [8][256]
  float* pl = ww + 8 * N_;             // [256]

  const int tid  = threadIdx.x;
  const int lane = tid & 31;
  const int w    = tid >> 5;
  const int b    = blockIdx.x >> 4;    // H = 16
  const int h    = blockIdx.x & 15;
  const size_t base = (size_t)b * N_;

  // stage K, V, policy for this (b,h) into LDS (read once from global)
  for (int idx = tid; idx < N_ * D_; idx += 256) {
    const int j = idx / D_;
    const int d = idx - j * D_;
    kt[j * KVP_ + d] = Kb[(base + j) * ld + h * D_ + d];
    vt[j * KVP_ + d] = Vb[(base + j) * ld + h * D_ + d];
  }
  pl[tid] = policy[base + tid];
  __syncthreads();

  for (int t = 0; t < N_ / 8; ++t) {
    const int i = t * 8 + w;           // this wave's query index
    const size_t tok = base + i;
    const float* qrow = Qb + tok * ld + h * D_;
    qw[w * 96 + lane]      = qrow[lane];
    qw[w * 96 + lane + 32] = qrow[lane + 32];
    if (lane < 24) qw[w * 96 + lane + 64] = qrow[lane + 64];
    __syncthreads();

    // scores: lane handles keys j = jo*32 + lane  (LDS reads conflict-free)
    float sc[8];
#pragma unroll
    for (int jo = 0; jo < 8; jo++) {
      const float* kr = kt + (jo * 32 + lane) * KVP_;
      float a = 0.f;
#pragma unroll 8
      for (int d = 0; d < D_; d++) a += qw[w * 96 + d] * kr[d];
      sc[jo] = a * SCALE_;
    }

    // row max over ALL keys (reference takes max before masking)
    float m = sc[0];
#pragma unroll
    for (int jo = 1; jo < 8; jo++) m = fmaxf(m, sc[jo]);
#pragma unroll
    for (int off = 16; off > 0; off >>= 1) m = fmaxf(m, __shfl_xor(m, off, 32));

    float aj[8];
    float ssum = 0.f;
#pragma unroll
    for (int jo = 0; jo < 8; jo++) {
      const int j = jo * 32 + lane;
      const float pol = pl[j];
      float ap;
      if (causal) ap = (j > i) ? 0.f : ((j == i) ? 1.f : pol);
      else        ap = 1.f - pol;
      const float e = __expf(sc[jo] - m) * ap;
      aj[jo] = e;
      ssum += e;
    }
#pragma unroll
    for (int off = 16; off > 0; off >>= 1) ssum += __shfl_xor(ssum, off, 32);

    const float inv = 1.0f / (ssum + EPS_);
#pragma unroll
    for (int jo = 0; jo < 8; jo++) ww[w * 256 + jo * 32 + lane] = (aj[jo] + EPSN_) * inv;
    __syncthreads();

    // out_d = sum_j w_j * V[j,d]  (V from LDS, lane-contiguous)
    float o0 = 0.f, o1 = 0.f, o2 = 0.f;
    for (int j = 0; j < N_; j++) {
      const float wj = ww[w * 256 + j];
      o0 += wj * vt[j * KVP_ + lane];
      o1 += wj * vt[j * KVP_ + lane + 32];
      if (lane < 24) o2 += wj * vt[j * KVP_ + lane + 64];
    }
    __bf16* op = outbf + tok * C_ + h * D_;
    op[lane]      = f2bf(o0);
    op[lane + 32] = f2bf(o1);
    if (lane < 24) op[lane + 64] = f2bf(o2);
    __syncthreads();
  }
}

// ---------------- host orchestration ----------------
extern "C" void kernel_launch(void* const* d_in, const int* in_sizes, int n_in,
                              void* d_out, int out_size, void* d_ws, size_t ws_size,
                              hipStream_t stream) {
  (void)in_sizes; (void)n_in; (void)out_size; (void)ws_size;
  const float* x           = (const float*)d_in[0];
  const float* decisions   = (const float*)d_in[1];
  const float* ln_vision_w = (const float*)d_in[2];
  const float* ln_vision_b = (const float*)d_in[3];
  const float* norm0_w     = (const float*)d_in[4];
  const float* norm0_b     = (const float*)d_in[5];
  const float* norm1_w     = (const float*)d_in[6];
  const float* norm1_b     = (const float*)d_in[7];
  const float* norm2_w     = (const float*)d_in[8];
  const float* norm2_b     = (const float*)d_in[9];
  const float* norm_w      = (const float*)d_in[10];
  const float* norm_b      = (const float*)d_in[11];
  const float* ca_qkv_w    = (const float*)d_in[12];
  const float* ca_qkv_b    = (const float*)d_in[13];
  const float* ca_proj_w   = (const float*)d_in[14];
  const float* ca_proj_b   = (const float*)d_in[15];
  const float* xa_q_w      = (const float*)d_in[16];
  const float* xa_q_b      = (const float*)d_in[17];
  const float* xa_k_w      = (const float*)d_in[18];
  const float* xa_k_b      = (const float*)d_in[19];
  const float* xa_v_w      = (const float*)d_in[20];
  const float* xa_v_b      = (const float*)d_in[21];
  const float* xa_proj_w   = (const float*)d_in[22];
  const float* xa_proj_b   = (const float*)d_in[23];
  const float* fc1_w       = (const float*)d_in[24];
  const float* fc1_b       = (const float*)d_in[25];
  const float* fc2_w       = (const float*)d_in[26];
  const float* fc2_b       = (const float*)d_in[27];

  // workspace carve-out (256B aligned)
  char* wsp = (char*)d_ws;
  auto alloc = [&](size_t bytes) -> char* {
    char* p = wsp;
    wsp += (bytes + 255) & ~(size_t)255;
    return p;
  };
  __bf16* qkvW    = (__bf16*)alloc((size_t)C3_ * C_ * 2);
  __bf16* caProjW = (__bf16*)alloc((size_t)C_ * C_ * 2);
  __bf16* xaQW    = (__bf16*)alloc((size_t)C_ * C_ * 2);
  __bf16* xaKW    = (__bf16*)alloc((size_t)C_ * C_ * 2);
  __bf16* xaVW    = (__bf16*)alloc((size_t)C_ * C_ * 2);
  __bf16* xaProjW = (__bf16*)alloc((size_t)C_ * C_ * 2);
  __bf16* fc1W    = (__bf16*)alloc((size_t)HID_ * C_ * 2);
  __bf16* fc2W    = (__bf16*)alloc((size_t)C_ * HID_ * 2);
  float*  xbuf    = (float*) alloc((size_t)T_ * C_ * 4);
  __bf16* originB = (__bf16*)alloc((size_t)T_ * C_ * 2);
  __bf16* actB    = (__bf16*)alloc((size_t)T_ * C_ * 2);
  char*   big     = alloc((size_t)T_ * C3_ * 4);   // 138.4 MB, reused
  float*  qkvF = (float*)big;
  float*  qF   = (float*)big;
  float*  kF   = qF + (size_t)T_ * C_;
  float*  vF   = kF + (size_t)T_ * C_;
  __bf16* hidB = (__bf16*)big;                     // 100.7 MB for [T, HID] bf16

  const dim3 blk(256);
  const dim3 gC3(C3_ / 128, T_ / 128);
  const dim3 gC (C_  / 128, T_ / 128);
  const dim3 gH (HID_ / 128, T_ / 128);
  const int cvtG = 2048;

  // 0) residual accumulator + bf16 weights
  copy_f32<<<4096, blk, 0, stream>>>(x, xbuf, (size_t)T_ * C_);
  cvt_f32_bf16<<<cvtG, blk, 0, stream>>>(ca_qkv_w,  qkvW,    (size_t)C3_ * C_);
  cvt_f32_bf16<<<cvtG, blk, 0, stream>>>(ca_proj_w, caProjW, (size_t)C_ * C_);
  cvt_f32_bf16<<<cvtG, blk, 0, stream>>>(xa_q_w,    xaQW,    (size_t)C_ * C_);
  cvt_f32_bf16<<<cvtG, blk, 0, stream>>>(xa_k_w,    xaKW,    (size_t)C_ * C_);
  cvt_f32_bf16<<<cvtG, blk, 0, stream>>>(xa_v_w,    xaVW,    (size_t)C_ * C_);
  cvt_f32_bf16<<<cvtG, blk, 0, stream>>>(xa_proj_w, xaProjW, (size_t)C_ * C_);
  cvt_f32_bf16<<<cvtG, blk, 0, stream>>>(fc1_w,     fc1W,    (size_t)HID_ * C_);
  cvt_f32_bf16<<<cvtG, blk, 0, stream>>>(fc2_w,     fc2W,    (size_t)C_ * HID_);

  // 1) x_origin = LN(x, ln_vision)  (bf16, feeds cross-attn K/V GEMMs)
  ln_kernel<true><<<T_, blk, 0, stream>>>(x, ln_vision_w, ln_vision_b, nullptr, originB);

  // 2) causal attention branch
  ln_kernel<true><<<T_, blk, 0, stream>>>(xbuf, norm0_w, norm0_b, nullptr, actB);
  gemm_bf16_wmma<0><<<gC3, blk, 0, stream>>>(actB, qkvW, ca_qkv_b, qkvF, nullptr, T_, C3_, C_);
  attn_kernel<<<B_ * H_, blk, ATTN_SMEM_BYTES_, stream>>>(qkvF, qkvF + C_, qkvF + 2 * C_,
                                                          C3_, decisions, actB, 1);
  gemm_bf16_wmma<1><<<gC, blk, 0, stream>>>(actB, caProjW, ca_proj_b, xbuf, nullptr, T_, C_, C_);

  // 3) cross attention branch
  ln_kernel<true><<<T_, blk, 0, stream>>>(xbuf, norm1_w, norm1_b, nullptr, actB);
  gemm_bf16_wmma<0><<<gC, blk, 0, stream>>>(actB,    xaQW, xa_q_b, qF, nullptr, T_, C_, C_);
  gemm_bf16_wmma<0><<<gC, blk, 0, stream>>>(originB, xaKW, xa_k_b, kF, nullptr, T_, C_, C_);
  gemm_bf16_wmma<0><<<gC, blk, 0, stream>>>(originB, xaVW, xa_v_b, vF, nullptr, T_, C_, C_);
  attn_kernel<<<B_ * H_, blk, ATTN_SMEM_BYTES_, stream>>>(qF, kF, vF, C_, decisions, actB, 0);
  gemm_bf16_wmma<1><<<gC, blk, 0, stream>>>(actB, xaProjW, xa_proj_b, xbuf, nullptr, T_, C_, C_);

  // 4) MLP with fused GELU->bf16 epilogue
  ln_kernel<true><<<T_, blk, 0, stream>>>(xbuf, norm2_w, norm2_b, nullptr, actB);
  gemm_bf16_wmma<2><<<gH, blk, 0, stream>>>(actB, fc1W, fc1_b, nullptr, hidB, T_, HID_, C_);
  gemm_bf16_wmma<1><<<gC, blk, 0, stream>>>(hidB, fc2W, fc2_b, xbuf, nullptr, T_, C_, HID_);

  // 5) final LN -> d_out (f32)
  ln_kernel<false><<<T_, blk, 0, stream>>>(xbuf, norm_w, norm_b, (float*)d_out, nullptr);
}
